// VectorQuantizer_54030688583900
// MI455X (gfx1250) — matrix operand order repaired and checked by reading, not compiled, not run on hardware
//
#include <hip/hip_runtime.h>
#include <hip/hip_bf16.h>

// ---------- types ----------
typedef __bf16 bf16_t;
typedef __attribute__((ext_vector_type(16))) __bf16 v16bf;
typedef __attribute__((ext_vector_type(8)))  __bf16 v8bf;
typedef __attribute__((ext_vector_type(8)))  float  v8f;

#define N_E    16384
#define E_DIM  256
#define N_ROWS 8192              // 8 * 32 * 32
#define HW     1024              // 32*32
#define NB     8
#define OUT_ELEMS 2097152        // 8*256*32*32
#define INV_SCALE (1.25f / 2097152.0f)

#define TILE_C   32              // codes staged per LDS tile
#define LDS_PAD  264             // 256 + 8 bf16 -> 528B row stride (bank-conflict free)
#define N_TILES  (N_E / TILE_C)  // 512

#define SHUF16(a, b) __builtin_shufflevector(a, b, 0,1,2,3,4,5,6,7,8,9,10,11,12,13,14,15)

// ---------- kernel 1: split embedding into bf16 hi/lo + exact ||e||^2 ----------
__global__ __launch_bounds__(256)
void vq_prep_embed(const float* __restrict__ emb, float* __restrict__ e2,
                   bf16_t* __restrict__ Ehi, bf16_t* __restrict__ Elo) {
    const int k = blockIdx.x;
    const int c = threadIdx.x;
    const float v = emb[k * E_DIM + c];
    const bf16_t h = (bf16_t)v;
    Ehi[k * E_DIM + c] = h;
    Elo[k * E_DIM + c] = (bf16_t)(v - (float)h);

    __shared__ float red[256];
    red[c] = v * v;
    __syncthreads();
    for (int s = 128; s > 0; s >>= 1) {
        if (c < s) red[c] += red[c + s];
        __syncthreads();
    }
    if (c == 0) e2[k] = red[0];
}

// ---------- kernel 2: transpose z (b,c,h,w)->(n,c) and split hi/lo ----------
__global__ __launch_bounds__(256)
void vq_prep_z(const float* __restrict__ z,
               bf16_t* __restrict__ Zhi, bf16_t* __restrict__ Zlo) {
    const int n = blockIdx.x;          // 0..8191
    const int c = threadIdx.x;         // 0..255
    const int b = n >> 10;
    const int hw = n & (HW - 1);
    const float v = z[((size_t)b * E_DIM + c) * HW + hw];
    const bf16_t h = (bf16_t)v;
    Zhi[n * E_DIM + c] = h;
    Zlo[n * E_DIM + c] = (bf16_t)(v - (float)h);
}

// ---------- kernel 3: zero loss slot ----------
__global__ void vq_zero_loss(float* loss) { *loss = 0.0f; }

// ---------- async copy: one 16B chunk per lane, global -> LDS (ASYNCcnt) ----------
__device__ __forceinline__ void async_b128_to_lds(bf16_t* lds_dst, const bf16_t* gsrc) {
    const unsigned int lds = (unsigned int)(uintptr_t)lds_dst;   // low 32 bits = LDS addr
    asm volatile("global_load_async_to_lds_b128 %0, %1, off"
                 :
                 : "v"(lds), "v"(gsrc)
                 : "memory");
}

// stage TILE_C codes (hi+lo) into an LDS buffer: 2048 b128 chunks / 128 threads = 16 each
__device__ __forceinline__ void stage_tile(const bf16_t* __restrict__ Ehi,
                                           const bf16_t* __restrict__ Elo,
                                           bf16_t* __restrict__ buf,   // [2][TILE_C][LDS_PAD]
                                           int c0, int tid) {
#pragma unroll
    for (int i = 0; i < 16; ++i) {
        const int id   = i * 128 + tid;       // 0..2047
        const int mat  = id >> 10;            // 0 = hi, 1 = lo
        const int rem  = id & 1023;
        const int code = rem >> 5;            // 0..31
        const int d16  = rem & 31;            // 16B chunk within the 512B row
        const bf16_t* g = (mat ? Elo : Ehi) + (size_t)(c0 + code) * E_DIM + d16 * 8;
        bf16_t* l = buf + mat * (TILE_C * LDS_PAD) + code * LDS_PAD + d16 * 8;
        async_b128_to_lds(l, g);
    }
}

// ---------- kernel 4: WMMA score sweep + fused argmin (LDS double-buffered) ----------
// Block = 128 threads = 4 waves; each wave owns a 16-row strip (64 rows/block).
__global__ __launch_bounds__(128)
void vq_argmin(const bf16_t* __restrict__ Zhi, const bf16_t* __restrict__ Zlo,
               const bf16_t* __restrict__ Ehi, const bf16_t* __restrict__ Elo,
               const float*  __restrict__ e2,  int* __restrict__ idx_out) {
    __shared__ bf16_t smem[2][2][TILE_C][LDS_PAD];   // [buf][hi/lo][code][dim] ~66KB

    const int tid  = threadIdx.x;
    const int wave = tid >> 5;
    const int lane = tid & 31;
    const int half = lane >> 4;       // 0 or 1
    const int l16  = lane & 15;
    const int strip = blockIdx.x * 64 + wave * 16;   // base row of this wave
    const int rowA  = strip + l16;                   // A-fragment row for this lane

    // ---- preload full A panel: 8 K-chunks of 32, hi & lo (contiguous k = 16h + s) ----
    v16bf Ahi[8], Alo[8];
    const bf16_t* zh = Zhi + (size_t)rowA * E_DIM;
    const bf16_t* zl = Zlo + (size_t)rowA * E_DIM;
#pragma unroll
    for (int kc = 0; kc < 8; ++kc) {
        const int kb = kc * 32;
        v8bf a0 = *(const v8bf*)(zh + kb + half * 16);       // K = h*16 .. h*16+7
        v8bf a1 = *(const v8bf*)(zh + kb + half * 16 + 8);   // K = h*16+8 ..
        Ahi[kc] = SHUF16(a0, a1);
        v8bf b0 = *(const v8bf*)(zl + kb + half * 16);
        v8bf b1 = *(const v8bf*)(zl + kb + half * 16 + 8);
        Alo[kc] = SHUF16(b0, b1);
    }

    float bestv[8];
    int   besti[8];
#pragma unroll
    for (int r = 0; r < 8; ++r) { bestv[r] = 3.4e38f; besti[r] = 0; }

    // prologue: stage tile 0
    stage_tile(Ehi, Elo, &smem[0][0][0][0], 0, tid);

    for (int t = 0; t < N_TILES; ++t) {
        const int c0  = t * TILE_C;
        const int cur = t & 1;
        if (t + 1 < N_TILES) {
            stage_tile(Ehi, Elo, &smem[cur ^ 1][0][0][0], c0 + TILE_C, tid);
            asm volatile("s_wait_asynccnt 16" ::: "memory");   // tile t complete
        } else {
            asm volatile("s_wait_asynccnt 0" ::: "memory");
        }
        __syncthreads();                                       // tile t visible to all waves

#pragma unroll
        for (int s = 0; s < 2; ++s) {                          // two 16-column subtiles
            const int code = c0 + s * 16 + l16;                // column this lane owns
            const float me2 = e2[code];                        // overlap with WMMA chain
            const bf16_t* eh = &smem[cur][0][s * 16 + l16][half * 16];
            const bf16_t* el = &smem[cur][1][s * 16 + l16][half * 16];
            v8f acc0 = {0,0,0,0,0,0,0,0};                      // hi*hi
            v8f acc1 = {0,0,0,0,0,0,0,0};                      // cross terms (ILP)

            // ---- software-pipelined K loop: prime chunk 0, prefetch k+1 ----
            v8bf h0 = *(const v8bf*)(eh + 0);
            v8bf h1 = *(const v8bf*)(eh + 8);
            v8bf g0 = *(const v8bf*)(el + 0);
            v8bf g1 = *(const v8bf*)(el + 8);
#pragma unroll
            for (int kc = 0; kc < 8; ++kc) {
                v16bf Bhi = SHUF16(h0, h1);
                v16bf Blo = SHUF16(g0, g1);
                if (kc < 7) {
                    const int kb = (kc + 1) * 32;
                    h0 = *(const v8bf*)(eh + kb);
                    h1 = *(const v8bf*)(eh + kb + 8);
                    g0 = *(const v8bf*)(el + kb);
                    g1 = *(const v8bf*)(el + kb + 8);
                }
                // Bhi consumers first: gives the Blo loads two extra issue slots
                acc0 = __builtin_amdgcn_wmma_f32_16x16x32_bf16(false, Ahi[kc], false, Bhi,
                                                               (short)0, acc0, false, false);
                acc1 = __builtin_amdgcn_wmma_f32_16x16x32_bf16(false, Alo[kc], false, Bhi,
                                                               (short)0, acc1, false, false);
                acc1 = __builtin_amdgcn_wmma_f32_16x16x32_bf16(false, Ahi[kc], false, Blo,
                                                               (short)0, acc1, false, false);
            }
#pragma unroll
            for (int r = 0; r < 8; ++r) {
                const float d = me2 - 2.0f * (acc0[r] + acc1[r]);
                if (d < bestv[r]) { bestv[r] = d; besti[r] = code; }
            }
        }
        __syncthreads();   // all waves done reading buf[cur] before tile t+2 overwrites it
    }

    // ---- cross-lane argmin within each 16-lane half (ties -> lowest index) ----
#pragma unroll
    for (int r = 0; r < 8; ++r) {
        float v = bestv[r];
        int   i = besti[r];
#pragma unroll
        for (int off = 1; off < 16; off <<= 1) {
            float ov = __shfl_xor(v, off, 32);
            int   oi = __shfl_xor(i, off, 32);
            if (ov < v || (ov == v && oi < i)) { v = ov; i = oi; }
        }
        // lanes 0-15 hold rows strip+0..7, lanes 16-31 hold rows strip+8..15
        if (l16 == 0) idx_out[strip + half * 8 + r] = i;
    }
}

// ---------- kernel 5: gather z_q to NCHW + fused loss ----------
__global__ __launch_bounds__(256)
void vq_output(const float* __restrict__ z, const float* __restrict__ emb,
               const int* __restrict__ idx, float* __restrict__ out,
               float* __restrict__ loss) {
    // blockIdx encodes (b, c, hw-tile): 8 * 256 * 4 = 8192 blocks
    const int bid = blockIdx.x;
    const int hwt = bid & 3;
    const int c   = (bid >> 2) & 255;
    const int b   = bid >> 10;
    const int hw  = hwt * 256 + threadIdx.x;
    const int n   = b * HW + hw;
    const int k   = idx[n];                       // coalesced over threads
    const float e  = emb[(size_t)k * E_DIM + c];  // gather (L2 resident)
    const size_t o = ((size_t)b * E_DIM + c) * HW + hw;
    const float zv = z[o];
    out[o] = e;                                   // straight-through forward == z_q
    const float d = e - zv;

    __shared__ float red[256];
    red[threadIdx.x] = d * d;
    __syncthreads();
    for (int s = 128; s > 0; s >>= 1) {
        if (threadIdx.x < s) red[threadIdx.x] += red[threadIdx.x + s];
        __syncthreads();
    }
    if (threadIdx.x == 0) atomicAdd(loss, red[0] * INV_SCALE);
}

// ---------- launch ----------
extern "C" void kernel_launch(void* const* d_in, const int* in_sizes, int n_in,
                              void* d_out, int out_size, void* d_ws, size_t ws_size,
                              hipStream_t stream) {
    const float* z   = (const float*)d_in[0];   // (8,256,32,32)
    const float* emb = (const float*)d_in[1];   // (16384,256)

    // d_out: [z_q_out: 2097152 f32][loss: 1 f32][idx: 8192 i32]
    float* out_q  = (float*)d_out;
    float* out_ls = out_q + OUT_ELEMS;
    int*   out_ix = (int*)d_out + OUT_ELEMS + 1;

    // workspace layout
    char* ws = (char*)d_ws;
    float*  e2  = (float*)ws;                                   //   64 KB
    bf16_t* Ehi = (bf16_t*)(ws + (size_t)65536);                //  8 MB
    bf16_t* Elo = (bf16_t*)(ws + (size_t)65536 + 8388608);      //  8 MB
    bf16_t* Zhi = (bf16_t*)(ws + (size_t)65536 + 2*8388608);    //  4 MB
    bf16_t* Zlo = (bf16_t*)(ws + (size_t)65536 + 2*8388608 + 4194304); // 4 MB

    vq_prep_embed<<<N_E, 256, 0, stream>>>(emb, e2, Ehi, Elo);
    vq_prep_z<<<N_ROWS, 256, 0, stream>>>(z, Zhi, Zlo);
    vq_zero_loss<<<1, 1, 0, stream>>>(out_ls);
    vq_argmin<<<N_ROWS / 64, 128, 0, stream>>>(Zhi, Zlo, Ehi, Elo, e2, out_ix);
    vq_output<<<NB * E_DIM * 4, 256, 0, stream>>>(z, emb, out_ix, out_q, out_ls);
}
